// PredictionScore_70789650973265
// MI455X (gfx1250) — matrix operand order repaired and checked by reading, not compiled, not run on hardware
//
#include <hip/hip_runtime.h>
#include <hip/hip_bf16.h>
#include <math.h>

typedef __attribute__((ext_vector_type(16))) _Float16 v16h;
typedef __attribute__((ext_vector_type(8)))  float    v8f;
typedef __attribute__((ext_vector_type(4)))  float    f4;

namespace {

constexpr int kB = 16, kNL = 48, kNP = 256, kD = 128;
constexpr int kNLD = kB * kNL;              // 768 dense ligand rows
constexpr int kNPD = kB * kNP;              // 4096 dense pocket rows
constexpr int kPairs = kB * kNL * kNP;      // 196608

// output sections (float element offsets), in reference return order
constexpr size_t OUT_PI    = 0;
constexpr size_t OUT_SIGMA = (size_t)kPairs * 10;
constexpr size_t OUT_MU    = (size_t)kPairs * 20;
constexpr size_t OUT_DIST  = (size_t)kPairs * 30;
constexpr size_t OUT_CM    = (size_t)kPairs * 31;

// workspace layout (byte offsets)
constexpr size_t WS_CUML = 0;                                  // int[17]
constexpr size_t WS_CUMP = 128;                                // int[17]
constexpr size_t WS_S    = 256;                                // float[128] BN scale
constexpr size_t WS_U    = 768;                                // float[128] BN/bias offset
constexpr size_t WS_BH   = 1280;                               // float[32] head biases
constexpr size_t WS_WL   = 4096;                               // f16 swz [4][8][32][16]
constexpr size_t WS_WP   = WS_WL + (size_t)4*8*32*16*2;        // 36864
constexpr size_t WS_WH   = WS_WP + (size_t)4*8*32*16*2;        // 69632, f16 swz [4][2][32][16]
constexpr size_t WS_PL   = WS_WH + (size_t)4*2*32*16*2;        // 77824, float [768][128]
constexpr size_t WS_PP   = WS_PL + (size_t)kNLD*kD*4;          // 471040, float [4096][128]
constexpr size_t WS_POSL = WS_PP + (size_t)kNPD*kD*4;          // 2568192, float [768][4]
constexpr size_t WS_POSP = WS_POSL + (size_t)kNLD*4*4;         // 2580480, float [4096][4]
constexpr size_t WS_ML   = WS_POSP + (size_t)kNPD*4*4;         // 2646016, int [768]
constexpr size_t WS_MP   = WS_ML + (size_t)kNLD*4;             // 2649088, int [4096]

__device__ __forceinline__ float eluf(float x) { return x > 0.f ? x : expm1f(x); }

__device__ __forceinline__ v8f wmma16(v16h a, v16h b, v8f c) {
  // D = A(16x32 f16) x B(32x16 f16) + C(16x16 f32)
  return __builtin_amdgcn_wmma_f32_16x16x32_f16(false, a, false, b, (short)0, c, false, false);
}

// ---------------------------------------------------------------------------
// Kernel 0: prefix sums, BN folding, head-bias gather, weight packing into the
// contiguous-k per-lane WMMA B-operand order matching A (lanes 0-15: K=e, lanes 16-31: K=16+e,
// column = lane&15). BN scale s[n] folded into W1 columns.
// ---------------------------------------------------------------------------
__global__ __launch_bounds__(256) void k_setup(
    const int* __restrict__ num_l, const int* __restrict__ num_p,
    const float* __restrict__ W1, const float* __restrict__ b1,
    const float* __restrict__ gma, const float* __restrict__ bta,
    const float* __restrict__ rmean, const float* __restrict__ rvar,
    const float* __restrict__ Wpi, const float* __restrict__ bpi,
    const float* __restrict__ Wsg, const float* __restrict__ bsg,
    const float* __restrict__ Wmu, const float* __restrict__ bmu,
    char* __restrict__ ws)
{
  const int tid = threadIdx.x;
  int*   cuml = (int*)(ws + WS_CUML);
  int*   cump = (int*)(ws + WS_CUMP);
  float* S  = (float*)(ws + WS_S);
  float* U  = (float*)(ws + WS_U);
  float* BH = (float*)(ws + WS_BH);
  _Float16* WL = (_Float16*)(ws + WS_WL);
  _Float16* WP = (_Float16*)(ws + WS_WP);
  _Float16* WH = (_Float16*)(ws + WS_WH);

  if (tid < kD) {
    float s = gma[tid] * rsqrtf(rvar[tid] + 1e-5f);
    S[tid] = s;
    U[tid] = (b1[tid] - rmean[tid]) * s + bta[tid];
  }
  if (tid >= 128 && tid < 160) {
    int j = tid - 128;
    BH[j] = (j < 10) ? bpi[j] : (j < 20) ? bsg[j - 10] : (j < 30) ? bmu[j - 20] : 0.f;
  }
  if (tid == 160) { int a = 0; cuml[0] = 0; for (int i = 0; i < kB; ++i) { a += num_l[i]; cuml[i+1] = a; } }
  if (tid == 161) { int a = 0; cump[0] = 0; for (int i = 0; i < kB; ++i) { a += num_p[i]; cump[i+1] = a; } }
  __syncthreads();

  // projection weights, BN scale folded in: 4 ktiles x 8 ntiles x 32 lanes x 16 halves
  for (int idx = tid; idx < 4*8*32*16; idx += 256) {
    const int e = idx & 15, lane = (idx >> 4) & 31, nt = (idx >> 9) & 7, kt = (idx >> 12) & 3;
    const int k = kt*32 + ((lane & 16) ? 16 : 0) + e;
    const int n = nt*16 + (lane & 15);
    const float s = S[n];
    WL[idx] = (_Float16)(W1[(size_t)k * kD + n] * s);
    WP[idx] = (_Float16)(W1[(size_t)(kD + k) * kD + n] * s);
  }
  // head weights [128 x 32]: cols 0-9 pi, 10-19 sigma, 20-29 mu, 30-31 zero
  for (int idx = tid; idx < 4*2*32*16; idx += 256) {
    const int e = idx & 15, lane = (idx >> 4) & 31, nt = (idx >> 9) & 1, kt = (idx >> 10) & 3;
    const int k = kt*32 + ((lane & 16) ? 16 : 0) + e;
    const int n = nt*16 + (lane & 15);
    float v = 0.f;
    if (n < 10)      v = Wpi[k*10 + n];
    else if (n < 20) v = Wsg[k*10 + (n - 10)];
    else if (n < 30) v = Wmu[k*10 + (n - 20)];
    WH[idx] = (_Float16)v;
  }
}

// ---------------------------------------------------------------------------
// Kernel 1: fused ragged->dense scatter + projection GEMM via WMMA.
// One wave = 16 dense rows x 128 features, K=128 -> 32 wmma per wave.
// Tiles 0..47 are L rows (adds BN offset u), 48..303 are P rows.
// Also writes dense positions (+|x|^2) and masks.
// ---------------------------------------------------------------------------
__global__ __launch_bounds__(128) void k_proj(
    const float* __restrict__ hl, const float* __restrict__ hp,
    const float* __restrict__ pos_l, const float* __restrict__ pos_p,
    const int* __restrict__ num_l, const int* __restrict__ num_p,
    char* __restrict__ ws)
{
  const int wave = threadIdx.x >> 5;
  const int lane = threadIdx.x & 31;
  const int tile = blockIdx.x * 4 + wave;
  const int m = lane & 15, hf = lane >> 4;
  const bool isL = tile < (kNLD / 16);

  const int*   cum  = (const int*)(ws + (isL ? WS_CUML : WS_CUMP));
  const int*   nums = isL ? num_l : num_p;
  const float* srcH = isL ? hl : hp;
  const float* spos = isL ? pos_l : pos_p;
  const _Float16* wsz = (const _Float16*)(ws + (isL ? WS_WL : WS_WP));
  float* dst  = (float*)(ws + (isL ? WS_PL : WS_PP));
  float* posd = (float*)(ws + (isL ? WS_POSL : WS_POSP));
  int*   mskd = (int*)(ws + (isL ? WS_ML : WS_MP));
  const float* U = (const float*)(ws + WS_U);
  const int Nmax = isL ? kNL : kNP;
  const int tile16 = (isL ? tile : tile - kNLD / 16) * 16;

  const int dr = tile16 + m;              // dense row this lane's A-row comes from
  const int bb = dr / Nmax;
  const int ii = dr - bb * Nmax;
  const bool valid = ii < nums[bb];
  const int srow = cum[bb] + ii;

  v8f acc[8] = {};
  #pragma unroll
  for (int kt = 0; kt < 4; ++kt) {
    v16h A = {};
    if (valid) {
      const float* rp = srcH + (size_t)srow * kD + kt*32 + hf*16;
      f4 x0 = *(const f4*)(rp);
      f4 x1 = *(const f4*)(rp + 4);
      f4 x2 = *(const f4*)(rp + 8);
      f4 x3 = *(const f4*)(rp + 12);
      #pragma unroll
      for (int e = 0; e < 4; ++e) {
        A[e]      = (_Float16)x0[e];
        A[4 + e]  = (_Float16)x1[e];
        A[8 + e]  = (_Float16)x2[e];
        A[12 + e] = (_Float16)x3[e];
      }
    }
    #pragma unroll
    for (int nt = 0; nt < 8; ++nt) {
      v16h Bv = *(const v16h*)(wsz + ((size_t)(kt*8 + nt) * 32 + lane) * 16);
      acc[nt] = wmma16(A, Bv, acc[nt]);
    }
  }

  const int n0 = lane & 15;
  #pragma unroll
  for (int nt = 0; nt < 8; ++nt) {
    const int f = nt*16 + n0;
    const float uf = isL ? U[f] : 0.f;   // BN offset folded onto L side only
    #pragma unroll
    for (int v = 0; v < 8; ++v) {
      const int r2 = tile16 + v + 8*hf;  // C/D layout: VGPR v -> M = v + 8*(lane/16)
      dst[(size_t)r2 * kD + f] = acc[nt][v] + uf;
    }
  }

  if (hf == 0) {  // one lane per dense row: positions + mask
    float x = 0.f, y = 0.f, z = 0.f;
    if (valid) { const float* q = spos + (size_t)srow * 3; x = q[0]; y = q[1]; z = q[2]; }
    float* pd = posd + (size_t)dr * 4;
    pd[0] = x; pd[1] = y; pd[2] = z; pd[3] = x*x + y*y + z*z;
    mskd[dr] = valid ? 1 : 0;
  }
}

// ---------------------------------------------------------------------------
// Kernel 2: fused pair kernel. One wave = 16 pairs (fixed l, 16 consecutive p).
// h = elu(proj_l + proj_p) built in registers (never hits HBM), head GEMM via
// 8 WMMAs (K=128, N=32), logits staged through per-wave LDS for the 10-way
// softmax; sigma/mu activations, distances, masks all fused.
// ---------------------------------------------------------------------------
__global__ __launch_bounds__(128) void k_main(const char* __restrict__ ws,
                                              float* __restrict__ out)
{
  __shared__ float smem[4 * 16 * 32];
  const int wave = threadIdx.x >> 5;
  const int lane = threadIdx.x & 31;
  const int t  = blockIdx.x * 4 + wave;       // wave-tile id, 0..12287
  const int pt = t & 15;                      // p-tile (16 pairs)
  const int l  = (t >> 4) % kNL;
  const int bb = t / (16 * kNL);
  const int m = lane & 15, hf = lane >> 4;

  const float* pl   = (const float*)(ws + WS_PL);
  const float* pp   = (const float*)(ws + WS_PP);
  const float* posl = (const float*)(ws + WS_POSL);
  const float* posp = (const float*)(ws + WS_POSP);
  const int* maskl  = (const int*)(ws + WS_ML);
  const int* maskp  = (const int*)(ws + WS_MP);
  const _Float16* wh = (const _Float16*)(ws + WS_WH);
  const float* BH   = (const float*)(ws + WS_BH);

  const int plrow = bb * kNL + l;
  const int pprow = bb * kNP + pt*16 + m;     // A-row m <-> pocket row
  const float* ap = pl + (size_t)plrow * kD;
  const float* bp = pp + (size_t)pprow * kD;

  v8f c0 = {}, c1 = {};
  #pragma unroll
  for (int kt = 0; kt < 4; ++kt) {
    const int kb = kt*32 + hf*16;
    f4 x0 = *(const f4*)(ap + kb)      + *(const f4*)(bp + kb);
    f4 x1 = *(const f4*)(ap + kb + 4)  + *(const f4*)(bp + kb + 4);
    f4 x2 = *(const f4*)(ap + kb + 8)  + *(const f4*)(bp + kb + 8);
    f4 x3 = *(const f4*)(ap + kb + 12) + *(const f4*)(bp + kb + 12);
    v16h A;
    #pragma unroll
    for (int e = 0; e < 4; ++e) {
      A[e]      = (_Float16)eluf(x0[e]);
      A[4 + e]  = (_Float16)eluf(x1[e]);
      A[8 + e]  = (_Float16)eluf(x2[e]);
      A[12 + e] = (_Float16)eluf(x3[e]);
    }
    v16h B0 = *(const v16h*)(wh + ((size_t)(kt*2 + 0) * 32 + lane) * 16);
    v16h B1 = *(const v16h*)(wh + ((size_t)(kt*2 + 1) * 32 + lane) * 16);
    c0 = wmma16(A, B0, c0);
    c1 = wmma16(A, B1, c1);
  }

  // stage 16x32 logit tile (+ head biases) into per-wave LDS
  float* L = smem + wave * 512;
  const int n0 = lane & 15;
  const float bh0 = BH[n0], bh1 = BH[16 + n0];
  #pragma unroll
  for (int v = 0; v < 8; ++v) {
    const int mr = v + 8*hf;
    L[mr*32 + n0]      = c0[v] + bh0;
    L[mr*32 + 16 + n0] = c1[v] + bh1;
  }
  __syncthreads();

  const int ml = maskl[plrow];
  const int p = pt*16 + m;
  const int prow = bb * kNP + p;
  const int cm = (ml != 0) && (maskp[prow] != 0);
  const float cmf = cm ? 1.f : 0.f;
  const size_t pair = (size_t)(bb * kNL + l) * kNP + p;
  const float* Lm = L + m * 32;

  if (hf == 0) {
    // softmax over the 10 pi logits of pair m
    float v[10], vmax = -3.4e38f;
    #pragma unroll
    for (int j = 0; j < 10; ++j) { v[j] = Lm[j]; vmax = fmaxf(vmax, v[j]); }
    float s = 0.f;
    #pragma unroll
    for (int j = 0; j < 10; ++j) { v[j] = __expf(v[j] - vmax); s += v[j]; }
    const float inv = cmf / s;
    float* op = out + OUT_PI + pair * 10;
    #pragma unroll
    for (int j = 0; j < 10; ++j) op[j] = v[j] * inv;
    // distance + mask
    f4 a4 = *(const f4*)(posl + (size_t)plrow * 4);
    f4 b4 = *(const f4*)(posp + (size_t)prow * 4);
    const float d2 = a4[3] + b4[3] - 2.f * (a4[0]*b4[0] + a4[1]*b4[1] + a4[2]*b4[2]);
    out[OUT_DIST + pair] = cmf * sqrtf(fmaxf(d2, 0.f));
    out[OUT_CM + pair]   = cmf;
  } else {
    float* os = out + OUT_SIGMA + pair * 10;
    float* om = out + OUT_MU + pair * 10;
    #pragma unroll
    for (int j = 0; j < 10; ++j) {
      os[j] = cmf * (eluf(Lm[10 + j]) + 1.1f);
      om[j] = cmf * (eluf(Lm[20 + j]) + 1.0f);
    }
  }
}

} // namespace

extern "C" void kernel_launch(void* const* d_in, const int* in_sizes, int n_in,
                              void* d_out, int out_size, void* d_ws, size_t ws_size,
                              hipStream_t stream) {
  (void)in_sizes; (void)n_in; (void)out_size; (void)ws_size;
  const float* hl    = (const float*)d_in[0];
  const float* hp    = (const float*)d_in[1];
  const float* pos_l = (const float*)d_in[2];
  const float* pos_p = (const float*)d_in[3];
  const int*   num_l = (const int*)d_in[4];
  const int*   num_p = (const int*)d_in[5];
  // d_in[6], d_in[7]: n_l_max / n_p_max (compile-time constants here)
  const float* W1    = (const float*)d_in[8];
  const float* b1    = (const float*)d_in[9];
  const float* gma   = (const float*)d_in[10];
  const float* bta   = (const float*)d_in[11];
  const float* rmean = (const float*)d_in[12];
  const float* rvar  = (const float*)d_in[13];
  const float* Wpi   = (const float*)d_in[14];
  const float* bpi   = (const float*)d_in[15];
  const float* Wsg   = (const float*)d_in[16];
  const float* bsg   = (const float*)d_in[17];
  const float* Wmu   = (const float*)d_in[18];
  const float* bmu   = (const float*)d_in[19];
  char* ws = (char*)d_ws;
  float* out = (float*)d_out;

  k_setup<<<1, 256, 0, stream>>>(num_l, num_p, W1, b1, gma, bta, rmean, rvar,
                                 Wpi, bpi, Wsg, bsg, Wmu, bmu, ws);
  // (768 + 4096) / 16 = 304 row-tiles, 4 waves per block
  k_proj<<<304 / 4, 128, 0, stream>>>(hl, hp, pos_l, pos_p, num_l, num_p, ws);
  // 196608 pairs / 16 per wave / 4 waves per block
  k_main<<<kPairs / 64, 128, 0, stream>>>(ws, out);
}
